// AutoregressiveLSTM_10015863734588
// MI455X (gfx1250) — compile-verified, matrix operand order checked by reading, not run on hardware
//
#include <hip/hip_runtime.h>
#include <hip/hip_bf16.h>

// ---------------------------------------------------------------------------
// AutoregressiveLSTM forward for MI455X (gfx1250, wave32, WMMA).
// Persistent-kernel 2-layer LSTM scan: ONE kernel runs all 127 timesteps with
// device-wide software barriers (2 per step) instead of 383 kernel launches.
// Phases are __noinline__ device functions so each gets its own register
// allocation / scheduling scope (fully-inlined variant spilled to scratch).
// Global pointers are explicitly cast to address_space(1) inside the callees
// so weight streams use global_load (LOADcnt only), not flat_load (which
// would also tick DScnt and serialize against the LDS staging stream).
// bf16 WMMA GEMMs with f32 accumulate, fused 4-gate cell, fused FC+softmax+NLL.
// ---------------------------------------------------------------------------

typedef __bf16 bf16;
typedef float        v8f   __attribute__((ext_vector_type(8)));
typedef bf16         v16bf __attribute__((ext_vector_type(16)));
typedef unsigned int v4u   __attribute__((ext_vector_type(4)));   // builtin 16B vector

#define B_  512
#define T_  128
#define V_  36
#define E_  128
#define H_  512
#define VP_ 48           // vocab padded to 3 WMMA n-tiles
#define G4_ 2048         // 4*H
#define NBLK_ 128        // persistent grid size

// d_out layout (floats): probs | emb | sample_loss | mean
#define PROBS_N  ((size_t)B_ * (T_ - 1) * V_)          // 2,340,864
#define EMB_N    ((size_t)B_ * T_ * E_)                // 8,388,608
#define SLOSS_N  ((size_t)B_)

// ---- explicit global (addrspace 1) pointer types; cast via integer ----
typedef __attribute__((address_space(1))) const v4u*   gc_v4u_p;
typedef __attribute__((address_space(1))) const float* gc_float_p;
typedef __attribute__((address_space(1))) float*       g_float_p;
typedef __attribute__((address_space(1))) bf16*        g_bf16_p;
typedef __attribute__((address_space(1))) const int*   gc_int_p;
#define TO_GLOBAL(T, p) ((T)(unsigned long long)(size_t)(p))

// ---------------------------------------------------------------------------
// Dynamic LDS accessor: inlined into each phase function so address-space
// inference sees the LDS origin locally -> ds_load/ds_store codegen.
// ---------------------------------------------------------------------------
__device__ __forceinline__ bf16* dyn_smem() {
  extern __shared__ bf16 smem[];
  return smem;
}

// ---------------------------------------------------------------------------
// WMMA fragment helpers.
// A (16x32 bf16) lane layout: row m = lane&15; K chunk base kb = (lane>>4)*8;
// lane holds K = kb..kb+7 and kb+16..kb+23  -> two contiguous 16B loads.
// B (32x16 bf16) is symmetric with n = lane&15; since we compute x @ W^T, the
// native row-major W[out][in] rows ARE the contiguous B-fragment rows.
// ---------------------------------------------------------------------------
__device__ __forceinline__ v16bf make_frag(const bf16* p) {   // LDS source
  v16bf v;
  v4u* u = reinterpret_cast<v4u*>(&v);
  u[0] = *reinterpret_cast<const v4u*>(p);
  u[1] = *reinterpret_cast<const v4u*>(p + 16);
  return v;
}

__device__ __forceinline__ v16bf make_frag_g(const bf16* p) { // global source
  gc_v4u_p q = TO_GLOBAL(gc_v4u_p, p);
  v16bf v;
  v4u* u = reinterpret_cast<v4u*>(&v);
  u[0] = q[0];
  u[1] = q[2];                    // +32 bytes == +16 bf16 elements
  return v;
}

__device__ __forceinline__ v8f wmma_bf16(v16bf a, v16bf b, v8f c) {
  return __builtin_amdgcn_wmma_f32_16x16x32_bf16(
      /*neg_a=*/false, a, /*neg_b=*/false, b,
      /*c_mod=*/(short)0, c, /*reuse_a=*/false, /*reuse_b=*/false);
}

__device__ __forceinline__ float sigm(float x) { return 1.0f / (1.0f + __expf(-x)); }

// ---------------------------------------------------------------------------
// Device-wide barrier (agent-scope acq/rel -> compiler emits L2-coherent
// cache controls). Counter+generation live in ws, zeroed every launch.
// All NBLK_ blocks are co-resident (64KB LDS -> 5 blocks/WGP, tiny grid).
// ---------------------------------------------------------------------------
__device__ __forceinline__ void grid_barrier(unsigned* cnt, unsigned* gen) {
  __syncthreads();
  if (threadIdx.x == 0) {
    unsigned g = __hip_atomic_load(gen, __ATOMIC_RELAXED, __HIP_MEMORY_SCOPE_AGENT);
    unsigned arrived =
        __hip_atomic_fetch_add(cnt, 1u, __ATOMIC_ACQ_REL, __HIP_MEMORY_SCOPE_AGENT);
    if (arrived == NBLK_ - 1) {
      __hip_atomic_store(cnt, 0u, __ATOMIC_RELAXED, __HIP_MEMORY_SCOPE_AGENT);
      __hip_atomic_fetch_add(gen, 1u, __ATOMIC_ACQ_REL, __HIP_MEMORY_SCOPE_AGENT);
    } else {
      while (__hip_atomic_load(gen, __ATOMIC_ACQUIRE, __HIP_MEMORY_SCOPE_AGENT) == g) {
        __builtin_amdgcn_s_sleep(1);
      }
    }
  }
  __syncthreads();
}

// ---------------------------------------------------------------------------
// Preparation kernels (run every launch; kernel_launch must be stateless).
// ---------------------------------------------------------------------------
__global__ void f32_to_bf16_kernel(const float* __restrict__ src,
                                   bf16* __restrict__ dst, int n) {
  int i = blockIdx.x * blockDim.x + threadIdx.x;
  if (i < n) dst[i] = (bf16)src[i];
}

// combined biases + padded FC weights/bias
__global__ void prep_misc_kernel(const float* __restrict__ bi0, const float* __restrict__ bh0,
                                 const float* __restrict__ bi1, const float* __restrict__ bh1,
                                 const float* __restrict__ Wfc, const float* __restrict__ bfc,
                                 float* __restrict__ bias0, float* __restrict__ bias1,
                                 bf16* __restrict__ WfcP, float* __restrict__ bfcP) {
  int i = blockIdx.x * blockDim.x + threadIdx.x;   // grid covers VP_*H_ = 24576
  if (i < G4_) { bias0[i] = bi0[i] + bh0[i]; bias1[i] = bi1[i] + bh1[i]; }
  if (i < VP_) bfcP[i] = (i < V_) ? bfc[i] : 0.0f;
  if (i < VP_ * H_) {
    int r = i / H_, k = i % H_;
    WfcP[i] = (bf16)((r < V_) ? Wfc[r * H_ + k] : 0.0f);
  }
}

// emb gather: f32 copy to d_out + time-major bf16 copy for layer-0 GEMM input
__global__ void embed_kernel(const int* __restrict__ x,
                             const float* __restrict__ table,
                             float* __restrict__ emb_out,
                             bf16* __restrict__ emb_bf) {
  size_t i = (size_t)blockIdx.x * blockDim.x + threadIdx.x;  // B*T*E
  if (i >= EMB_N) return;
  int e = (int)(i & (E_ - 1));
  size_t bt = i >> 7;                 // E_=128
  int t = (int)(bt & (T_ - 1));
  int b = (int)(bt >> 7);             // T_=128
  float v = table[(size_t)x[bt] * E_ + e];
  emb_out[i] = v;
  if (t < T_ - 1)
    emb_bf[(((size_t)t * B_) + b) * E_ + e] = (bf16)v;
}

// ---------------------------------------------------------------------------
// One LSTM layer (device-side phase of the persistent kernel, NOINLINE):
//   gates[512,2048] = A1[512,K1] @ W1[2048,K1]^T + A2[512,512] @ W2[2048,512]^T + bias
//   cell update (i,f,g,o) -> c (f32, in-place), h_out (bf16, double-buffered)
// Block = 32 rows (two 16-row M-tiles) x 4 n-tiles (64 h-columns), 4 waves.
// Wave = both M-tiles x 1 n-tile x all 4 gates:
//   per K-chunk: 2 LDS A-frags + 4 global B-frags -> 8 WMMAs.
// ---------------------------------------------------------------------------
template <int K1>
__device__ __attribute__((noinline)) void lstm_layer_dev(
    const bf16* __restrict__ A1, const bf16* __restrict__ W1,
    const bf16* __restrict__ A2, const bf16* __restrict__ W2,
    const float* __restrict__ bias,
    float* __restrict__ c, bf16* __restrict__ h_out) {
  bf16* shA1 = dyn_smem();
  bf16* shA2 = shA1 + 32 * K1;

  const int m0 = (blockIdx.x >> 3) * 32;            // 16 m-blocks of 32 rows
  const int wave = threadIdx.x >> 5;
  const int lane = threadIdx.x & 31;
  const int ntile = (blockIdx.x & 7) * 4 + wave;    // 32 n-tiles of 16 cols

  __syncthreads();  // smem reused across phases: prior readers must be done
  // cooperative staging of the 32 activation rows into LDS (16B vectors)
  {
    const int k1v = K1 >> 3;                        // 16B chunks per A1 row
    v4u* dst1 = reinterpret_cast<v4u*>(shA1);
    for (int i = threadIdx.x; i < 32 * k1v; i += 128) {
      int r = i / k1v, cb = i % k1v;
      dst1[i] = TO_GLOBAL(gc_v4u_p, A1 + (size_t)(m0 + r) * K1)[cb];
    }
    v4u* dst2 = reinterpret_cast<v4u*>(shA2);
    for (int i = threadIdx.x; i < 32 * 64; i += 128) {   // 512/8 = 64 per row
      int r = i >> 6, cb = i & 63;
      dst2[i] = TO_GLOBAL(gc_v4u_p, A2 + (size_t)(m0 + r) * H_)[cb];
    }
  }
  __syncthreads();

  const int mrow = lane & 15;
  const int kb   = (lane >> 4) * 8;
  const int ncol = ntile * 16 + (lane & 15);

  // acc[mt][gate]
  v8f acc[2][4] = {{{}, {}, {}, {}}, {{}, {}, {}, {}}};

  // ---- input GEMM: K = K1 ----
  {
    const size_t gs = (size_t)H_ * K1;              // gate stride in W1 rows
    const bf16* wrow = W1 + (size_t)ncol * K1;
    for (int kc = 0; kc < K1; kc += 32) {
      v16bf a0 = make_frag(shA1 + (0 * 16 + mrow) * K1 + kc + kb);
      v16bf a1 = make_frag(shA1 + (1 * 16 + mrow) * K1 + kc + kb);
      const bf16* wp = wrow + kc + kb;
#pragma unroll
      for (int g = 0; g < 4; ++g) {
        v16bf b = make_frag_g(wp);
        acc[0][g] = wmma_bf16(a0, b, acc[0][g]);
        acc[1][g] = wmma_bf16(a1, b, acc[1][g]);
        wp += gs;
      }
    }
  }
  // ---- recurrent GEMM: K = 512 ----
  {
    const size_t gs = (size_t)H_ * H_;
    const bf16* wrow = W2 + (size_t)ncol * H_;
    for (int kc = 0; kc < H_; kc += 32) {
      v16bf a0 = make_frag(shA2 + (0 * 16 + mrow) * H_ + kc + kb);
      v16bf a1 = make_frag(shA2 + (1 * 16 + mrow) * H_ + kc + kb);
      const bf16* wp = wrow + kc + kb;
#pragma unroll
      for (int g = 0; g < 4; ++g) {
        v16bf b = make_frag_g(wp);
        acc[0][g] = wmma_bf16(a0, b, acc[0][g]);
        acc[1][g] = wmma_bf16(a1, b, acc[1][g]);
        wp += gs;
      }
    }
  }

  // ---- fused LSTM cell: PyTorch gate order i, f, g, o ----
  gc_float_p bg = TO_GLOBAL(gc_float_p, bias);
  g_float_p  cg = TO_GLOBAL(g_float_p, c);
  g_bf16_p   hg = TO_GLOBAL(g_bf16_p, h_out);
  const float b_i = bg[0 * H_ + ncol];
  const float b_f = bg[1 * H_ + ncol];
  const float b_g = bg[2 * H_ + ncol];
  const float b_o = bg[3 * H_ + ncol];
  const int mhi = (lane >> 4) * 8;                  // C/D layout: VGPR r -> M = r (+8 hi half)
#pragma unroll
  for (int mt = 0; mt < 2; ++mt) {
#pragma unroll
    for (int r = 0; r < 8; ++r) {
      const int m = m0 + mt * 16 + r + mhi;
      const size_t idx = (size_t)m * H_ + ncol;
      float iv = sigm(acc[mt][0][r] + b_i);
      float fv = sigm(acc[mt][1][r] + b_f);
      float gv = tanhf(acc[mt][2][r] + b_g);
      float ov = sigm(acc[mt][3][r] + b_o);
      float cn = fv * cg[idx] + iv * gv;
      cg[idx] = cn;
      hg[idx] = (bf16)(ov * tanhf(cn));
    }
  }
}

// ---------------------------------------------------------------------------
// FC head phase (blocks 0..7 only, NOINLINE): logits = relu(h1 @ Wfc^T + bfc)
// via WMMA, per-row softmax over V_=36 -> probs, NLL accumulated.
// Block = 64 rows, 4 waves; logits buffer overlays the staging LDS.
// Two-pass softmax (re-reads LDS) to keep register pressure low.
// ---------------------------------------------------------------------------
__device__ __attribute__((noinline)) void fc_softmax_dev(
    const bf16* __restrict__ h1, const bf16* __restrict__ Wfc,
    const float* __restrict__ bfc, const int* __restrict__ x,
    int t, float* __restrict__ probs_out, float* __restrict__ loss_acc) {
  float* lg = reinterpret_cast<float*>(dyn_smem()); // [64][VP_] floats = 12KB
  const int wave = threadIdx.x >> 5;
  const int lane = threadIdx.x & 31;
  const int mtile = blockIdx.x * 4 + wave;
  const int mrow = lane & 15;
  const int kb = (lane >> 4) * 8;

  v8f acc[3] = {{}, {}, {}};
  const bf16* arow = h1 + (size_t)(mtile * 16 + mrow) * H_;
  for (int kc = 0; kc < H_; kc += 32) {
    v16bf a = make_frag_g(arow + kc + kb);
#pragma unroll
    for (int nt = 0; nt < 3; ++nt) {
      const bf16* wp = Wfc + (size_t)(nt * 16 + (lane & 15)) * H_ + kc + kb;
      acc[nt] = wmma_bf16(a, make_frag_g(wp), acc[nt]);
    }
  }

  gc_float_p bfg = TO_GLOBAL(gc_float_p, bfc);
  const int mhi = (lane >> 4) * 8;
#pragma unroll
  for (int nt = 0; nt < 3; ++nt) {
    const int ncol = nt * 16 + (lane & 15);
    const float bv = bfg[ncol];
#pragma unroll
    for (int r = 0; r < 8; ++r) {
      float v = acc[nt][r] + bv;
      lg[(wave * 16 + r + mhi) * VP_ + ncol] = v > 0.0f ? v : 0.0f;  // relu BEFORE softmax
    }
  }
  __syncthreads();

  if (threadIdx.x < 64) {
    const int r = threadIdx.x;
    const int b = blockIdx.x * 64 + r;
    const float* row = lg + r * VP_;
    float mx = row[0];
    for (int v = 1; v < V_; ++v) mx = fmaxf(mx, row[v]);
    float s = 0.0f;
    for (int v = 0; v < V_; ++v) s += __expf(row[v] - mx);
    const float inv = 1.0f / s;
    g_float_p po = TO_GLOBAL(g_float_p, probs_out + ((size_t)b * (T_ - 1) + t) * V_);
    for (int v = 0; v < V_; ++v) po[v] = __expf(row[v] - mx) * inv;
    const int tgt = TO_GLOBAL(gc_int_p, x)[(size_t)b * T_ + t + 1];
    g_float_p la = TO_GLOBAL(g_float_p, loss_acc);
    la[b] += (tgt == 0) ? 0.0f : -row[tgt];                  // ignore_index = 0
  }
}

// ---------------------------------------------------------------------------
// Persistent scan kernel: all 127 timesteps, 2 grid barriers per step.
//   L0 -> barrier -> L1 -> barrier -> FC (blocks 0..7) -> next t
// FC needs no trailing barrier: it only reads h1[1-p] and its own loss rows,
// while L0(t+1) touches only h0/c0; per-block program order covers the rest.
// ---------------------------------------------------------------------------
__global__ __launch_bounds__(128)
void lstm_scan_kernel(const bf16* __restrict__ embbf,
                      const bf16* __restrict__ Wih0, const bf16* __restrict__ Whh0,
                      const bf16* __restrict__ Wih1, const bf16* __restrict__ Whh1,
                      const bf16* __restrict__ WfcP,
                      const float* __restrict__ bias0, const float* __restrict__ bias1,
                      const float* __restrict__ bfcP, const int* __restrict__ x,
                      float* __restrict__ c0, float* __restrict__ c1,
                      bf16* __restrict__ h0buf, bf16* __restrict__ h1buf,
                      float* __restrict__ probs, float* __restrict__ lossa,
                      unsigned* __restrict__ barcnt, unsigned* __restrict__ bargen) {
#pragma clang loop unroll(disable)
  for (int t = 0; t < T_ - 1; ++t) {
    const int p = t & 1;
    const bf16* xt = embbf + (size_t)t * B_ * E_;
    bf16* h0_prev = h0buf + (size_t)p * B_ * H_;
    bf16* h0_next = h0buf + (size_t)(1 - p) * B_ * H_;
    bf16* h1_prev = h1buf + (size_t)p * B_ * H_;
    bf16* h1_next = h1buf + (size_t)(1 - p) * B_ * H_;

    lstm_layer_dev<E_>(xt, Wih0, h0_prev, Whh0, bias0, c0, h0_next);
    grid_barrier(barcnt, bargen);
    lstm_layer_dev<H_>(h0_next, Wih1, h1_prev, Whh1, bias1, c1, h1_next);
    grid_barrier(barcnt, bargen);
    if (blockIdx.x < 8)
      fc_softmax_dev(h1_next, WfcP, bfcP, x, t, probs, lossa);
  }
}

// ---------------------------------------------------------------------------
// sample_loss = loss_sum / count(x != 0); mean over B.
// ---------------------------------------------------------------------------
__global__ __launch_bounds__(512)
void finalize_kernel(const int* __restrict__ x, const float* __restrict__ loss_acc,
                     float* __restrict__ sloss, float* __restrict__ meanp) {
  __shared__ float red[512];
  const int b = threadIdx.x;
  int len = 0;
  for (int t = 0; t < T_; ++t) len += (x[(size_t)b * T_ + t] != 0);
  const float sl = loss_acc[b] / (float)len;
  sloss[b] = sl;
  red[b] = sl;
  __syncthreads();
  for (int s = 256; s >= 1; s >>= 1) {
    if (b < s) red[b] += red[b + s];
    __syncthreads();
  }
  if (b == 0) *meanp = red[0] / (float)B_;
}

// ---------------------------------------------------------------------------
// Host launcher. Stateless: all prep redone each call; graph-capture safe
// (only kernel launches + hipMemsetAsync on `stream`). 8 launches total.
// ---------------------------------------------------------------------------
extern "C" void kernel_launch(void* const* d_in, const int* in_sizes, int n_in,
                              void* d_out, int out_size, void* d_ws, size_t ws_size,
                              hipStream_t stream) {
  (void)in_sizes; (void)n_in; (void)out_size; (void)ws_size;

  const int*   x      = (const int*)d_in[0];
  const float* table  = (const float*)d_in[1];
  const float* W_ih0  = (const float*)d_in[2];
  const float* W_hh0  = (const float*)d_in[3];
  const float* b_ih0  = (const float*)d_in[4];
  const float* b_hh0  = (const float*)d_in[5];
  const float* W_ih1  = (const float*)d_in[6];
  const float* W_hh1  = (const float*)d_in[7];
  const float* b_ih1  = (const float*)d_in[8];
  const float* b_hh1  = (const float*)d_in[9];
  const float* W_fc   = (const float*)d_in[10];
  const float* b_fc   = (const float*)d_in[11];

  float* out     = (float*)d_out;
  float* probs   = out;
  float* emb_out = out + PROBS_N;
  float* sloss   = out + PROBS_N + EMB_N;
  float* meanp   = out + PROBS_N + EMB_N + SLOSS_N;

  // ---- workspace carve (256B aligned) ----
  size_t off = 0;
  char* base = (char*)d_ws;
  auto carve = [&](size_t bytes) -> char* {
    char* p = base + off;
    off += (bytes + 255) & ~(size_t)255;
    return p;
  };
  bf16*  Wih0b  = (bf16*)carve((size_t)G4_ * E_ * 2);
  bf16*  Whh0b  = (bf16*)carve((size_t)G4_ * H_ * 2);
  bf16*  Wih1b  = (bf16*)carve((size_t)G4_ * H_ * 2);
  bf16*  Whh1b  = (bf16*)carve((size_t)G4_ * H_ * 2);
  bf16*  WfcP   = (bf16*)carve((size_t)VP_ * H_ * 2);
  float* bias0  = (float*)carve((size_t)G4_ * 4);
  float* bias1  = (float*)carve((size_t)G4_ * 4);
  float* bfcP   = (float*)carve((size_t)VP_ * 4);
  char*  states = base + off;                 // contiguous zero-init region:
  bf16*  h0buf  = (bf16*)carve((size_t)2 * B_ * H_ * 2);   //  h0 double buffer
  bf16*  h1buf  = (bf16*)carve((size_t)2 * B_ * H_ * 2);   //  h1 double buffer
  float* c0     = (float*)carve((size_t)B_ * H_ * 4);
  float* c1     = (float*)carve((size_t)B_ * H_ * 4);
  float* lossa  = (float*)carve((size_t)B_ * 4);
  unsigned* barcnt = (unsigned*)carve(256);                //  barrier counter
  unsigned* bargen = (unsigned*)carve(256);                //  barrier generation
  size_t states_bytes = (size_t)((base + off) - states);
  bf16*  embbf  = (bf16*)carve((size_t)(T_ - 1) * B_ * E_ * 2);

  // ---- init states + barrier vars (re-zeroed every call: stateless) ----
  (void)hipMemsetAsync(states, 0, states_bytes, stream);

  // ---- weight prep ----
  {
    int n;
    n = G4_ * E_;
    f32_to_bf16_kernel<<<(n + 255) / 256, 256, 0, stream>>>(W_ih0, Wih0b, n);
    n = G4_ * H_;
    f32_to_bf16_kernel<<<(n + 255) / 256, 256, 0, stream>>>(W_hh0, Whh0b, n);
    f32_to_bf16_kernel<<<(n + 255) / 256, 256, 0, stream>>>(W_ih1, Wih1b, n);
    f32_to_bf16_kernel<<<(n + 255) / 256, 256, 0, stream>>>(W_hh1, Whh1b, n);
    n = VP_ * H_;
    prep_misc_kernel<<<(n + 255) / 256, 256, 0, stream>>>(
        b_ih0, b_hh0, b_ih1, b_hh1, W_fc, b_fc, bias0, bias1, WfcP, bfcP);
  }

  // ---- embedding gather (f32 output + time-major bf16 GEMM input) ----
  embed_kernel<<<(unsigned)((EMB_N + 255) / 256), 256, 0, stream>>>(
      x, table, emb_out, embbf);

  // ---- persistent sequential LSTM scan (single launch, 127 steps) ----
  const size_t smem_bytes = (size_t)32 * (H_ + H_) * 2;   // 64 KB
  lstm_scan_kernel<<<NBLK_, 128, smem_bytes, stream>>>(
      embbf, Wih0b, Whh0b, Wih1b, Whh1b, WfcP, bias0, bias1, bfcP, x,
      c0, c1, h0buf, h1buf, probs, lossa, barcnt, bargen);

  // ---- epilogue ----
  finalize_kernel<<<1, 512, 0, stream>>>(x, lossa, sloss, meanp);
}
